// MultiHeadDotProductAttention_29265907155165
// MI455X (gfx1250) — matrix operand order repaired and checked by reading, not compiled, not run on hardware
//
#include <hip/hip_runtime.h>

// ---------------------------------------------------------------------------
// MI455X (gfx1250) fused MHA forward, bf16 WMMA everywhere.
//   B=4, S=2048, D_MODEL=1024, H=16, D=64
// GEMMs: 256x64 C-tile per block, 8 waves x (32 rows = 2 strips), K-step 64.
//   Each B-fragment feeds 2 WMMAs (strip reuse): ~1.5 ds_load_b128 / WMMA,
//   16 WMMAs per barrier pair.
// Workspace (~88 MB): xb[8192,1024]bf16, W^T x4 [1024,1024]bf16,
//   qb,kb [B,H,S,D]bf16, vtb [B,H,D,S]bf16, cc [B,S,1024]bf16
// ---------------------------------------------------------------------------

typedef __attribute__((ext_vector_type(16))) __bf16 v16bf;
typedef __attribute__((ext_vector_type(4)))  __bf16 v4bf;
typedef __attribute__((ext_vector_type(8)))  float  v8f;

union FragB { v16bf v; uint4 u[2]; };

__device__ __forceinline__ float red_max16(float v) {
#pragma unroll
  for (int m = 1; m < 16; m <<= 1) v = fmaxf(v, __shfl_xor(v, m));
  return v;
}
__device__ __forceinline__ float red_sum16(float v) {
#pragma unroll
  for (int m = 1; m < 16; m <<= 1) v += __shfl_xor(v, m);
  return v;
}

// ---------------- conversion kernels ----------------

__global__ void cvt_f32_to_bf16(const float* __restrict__ in,
                                __bf16* __restrict__ out, int n) {
  int i = (blockIdx.x * blockDim.x + threadIdx.x) * 4;
  if (i < n) {
    float4 f = *(const float4*)(in + i);
    v4bf o;
    o.x = (__bf16)f.x; o.y = (__bf16)f.y; o.z = (__bf16)f.z; o.w = (__bf16)f.w;
    *(v4bf*)(out + i) = o;
  }
}

// out[n][k] = (bf16) in[k][n], 1024x1024
__global__ void transpose_cvt_bf16(const float* __restrict__ in,
                                   __bf16* __restrict__ out) {
  __shared__ float tile[16][17];
  int bx = blockIdx.x * 16, by = blockIdx.y * 16;
  int tx = threadIdx.x, ty = threadIdx.y;
  tile[ty][tx] = in[(size_t)(by + ty) * 1024 + bx + tx];
  __syncthreads();
  out[(size_t)(bx + ty) * 1024 + by + tx] = (__bf16)tile[tx][ty];
}

// ---------------- fused projection GEMM + per-head LayerNorm ----------------
// grid=(32,16): x = 256-row tile, y = head (64-col tile == one head).
// vt_mode==0 -> out[B,H,S,D]; vt_mode==1 -> out[B,H,D,S]
__global__ __launch_bounds__(256)
void proj_gemm_ln(const __bf16* __restrict__ A, const __bf16* __restrict__ Wt,
                  const float* __restrict__ bias, __bf16* __restrict__ out,
                  int vt_mode) {
  __shared__ __bf16 At[256 * 64];  // 32 KB
  __shared__ __bf16 Bt[64 * 64];   // 8 KB
  const int t = threadIdx.x;
  const int wv = t >> 5, lane = t & 31, lane15 = lane & 15, hi = lane >> 4;
  const int row0 = blockIdx.x * 256;
  const int h = blockIdx.y;
  const int n0 = h * 64;

  v8f acc[2][4] = {};  // [strip][nb]

  for (int k0 = 0; k0 < 1024; k0 += 64) {
    __syncthreads();
    {  // A tile: 256 rows x 64 k; one full 128B row per thread
      const __bf16* g = A + (size_t)(row0 + t) * 1024 + k0;
      __bf16* d = &At[t * 64];
#pragma unroll
      for (int c = 0; c < 64; c += 8) *(uint4*)(d + c) = *(const uint4*)(g + c);
    }
    {  // B tile: Bt[n][k], 64 x 64, 32B per thread
      int r = t >> 2, c = (t & 3) * 16;
      const __bf16* g = Wt + (size_t)(n0 + r) * 1024 + k0 + c;
      __bf16* d = &Bt[r * 64 + c];
      *(uint4*)(d)     = *(const uint4*)(g);
      *(uint4*)(d + 8) = *(const uint4*)(g + 8);
    }
    __syncthreads();

    // A fragments for both strips (batched), then per-N-block B pair reused
    // by both strips: each B fragment feeds 2 WMMAs.
    FragB a[2][2];  // [strip][h2]
#pragma unroll
    for (int s = 0; s < 2; ++s) {
      const int ar = wv * 32 + s * 16 + lane15;
#pragma unroll
      for (int h2 = 0; h2 < 2; ++h2) {
        const __bf16* p = &At[ar * 64 + h2 * 32];
        a[s][h2].u[0] = *(const uint4*)(p + hi * 8);
        a[s][h2].u[1] = *(const uint4*)(p + 16 + hi * 8);
      }
    }
#pragma unroll
    for (int nb = 0; nb < 4; ++nb) {
      FragB b[2];
#pragma unroll
      for (int h2 = 0; h2 < 2; ++h2) {
        const __bf16* p = &Bt[(nb * 16 + lane15) * 64 + h2 * 32 + hi * 8];
        b[h2].u[0] = *(const uint4*)p;
        b[h2].u[1] = *(const uint4*)(p + 16);
      }
#pragma unroll
      for (int s = 0; s < 2; ++s)
#pragma unroll
        for (int h2 = 0; h2 < 2; ++h2)
          acc[s][nb] = __builtin_amdgcn_wmma_f32_16x16x32_bf16(
              false, a[s][h2].v, false, b[h2].v, (short)0, acc[s][nb],
              false, false);
    }
  }

  // Epilogue: +bias, LayerNorm over the 64-wide head, fully in-register.
  float bv[4];
#pragma unroll
  for (int nb = 0; nb < 4; ++nb) bv[nb] = bias[n0 + nb * 16 + lane15];

#pragma unroll
  for (int s = 0; s < 2; ++s) {
#pragma unroll
    for (int i = 0; i < 8; ++i) {  // C row = i + hi*8
      float v0 = acc[s][0][i] + bv[0];
      float v1 = acc[s][1][i] + bv[1];
      float v2 = acc[s][2][i] + bv[2];
      float v3 = acc[s][3][i] + bv[3];
      float s1 = red_sum16(v0 + v1 + v2 + v3);
      float s2 = red_sum16(v0 * v0 + v1 * v1 + v2 * v2 + v3 * v3);
      float mean = s1 * (1.0f / 64.0f);
      float var  = s2 * (1.0f / 64.0f) - mean * mean;
      float rstd = rsqrtf(var + 1e-5f);
      float w4[4] = {(v0 - mean) * rstd, (v1 - mean) * rstd,
                     (v2 - mean) * rstd, (v3 - mean) * rstd};

      int r = row0 + wv * 32 + s * 16 + i + hi * 8;
      int bsel = r >> 11, srow = r & 2047;
      size_t bh = (size_t)bsel * 16 + h;
#pragma unroll
      for (int nb = 0; nb < 4; ++nb) {
        int d = nb * 16 + lane15;
        if (vt_mode)
          out[(bh * 64 + d) * 2048 + srow] = (__bf16)w4[nb];
        else
          out[(bh * 2048 + srow) * 64 + d] = (__bf16)w4[nb];
      }
    }
  }
}

// ---------------- flash attention: 64-key tiles, 16 WMMAs / 3 barriers ------
// grid = (S/128, B*H), block = 256 (8 waves, 16 q-rows each).
__global__ __launch_bounds__(256)
void flash_attn(const __bf16* __restrict__ q, const __bf16* __restrict__ k,
                const __bf16* __restrict__ vt, __bf16* __restrict__ concat) {
  __shared__ __bf16 Kt[64 * 64];      // [key][d]   8 KB
  __shared__ __bf16 Vt[64 * 64];      // [d][key]   8 KB
  __shared__ __bf16 Pt[8 * 16 * 64];  // per-wave P [16 q][64 key]  16 KB

  const int t = threadIdx.x;
  const int wv = t >> 5, lane = t & 31, lane15 = lane & 15, hi = lane >> 4;
  const int bh = blockIdx.y;
  const int q0 = blockIdx.x * 128 + wv * 16;

  const __bf16* qb = q  + (size_t)bh * 2048 * 64;
  const __bf16* kb = k  + (size_t)bh * 2048 * 64;
  const __bf16* vb = vt + (size_t)bh * 64 * 2048;

  // Q A-fragments for d=[0,32) and d=[32,64), loaded once.
  FragB aq[2];
  {
    const __bf16* qr = qb + (size_t)(q0 + lane15) * 64;
#pragma unroll
    for (int h2 = 0; h2 < 2; ++h2) {
      aq[h2].u[0] = *(const uint4*)(qr + h2 * 32 + hi * 8);
      aq[h2].u[1] = *(const uint4*)(qr + h2 * 32 + 16 + hi * 8);
    }
  }

  v8f o[4] = {};
  float M[8], l[8];
#pragma unroll
  for (int i = 0; i < 8; ++i) { M[i] = -3.0e38f; l[i] = 0.0f; }

  __bf16* Pw = &Pt[wv * 16 * 64];
  const float scale = 0.125f;  // 1/sqrt(64)

  for (int key0 = 0; key0 < 2048; key0 += 64) {
    __syncthreads();  // previous tile fully consumed
    {  // K tile 64x64: one 128B row per 2 threads
      int r = t >> 1, c = (t & 1) * 32;
      const __bf16* g = kb + (size_t)(key0 + r) * 64 + c;
      __bf16* d = &Kt[r * 64 + c];
      *(uint4*)(d)      = *(const uint4*)(g);
      *(uint4*)(d + 8)  = *(const uint4*)(g + 8);
      *(uint4*)(d + 16) = *(const uint4*)(g + 16);
      *(uint4*)(d + 24) = *(const uint4*)(g + 24);
    }
    {  // V^T tile 64 d x 64 keys
      int r = t >> 2, c = (t & 3) * 16;
      const __bf16* g = vb + (size_t)r * 2048 + key0 + c;
      __bf16* d = &Vt[r * 64 + c];
      *(uint4*)(d)     = *(const uint4*)(g);
      *(uint4*)(d + 8) = *(const uint4*)(g + 8);
    }
    if (key0 + 64 < 2048)  // -> global_prefetch_b8 of next K tile
      __builtin_prefetch(kb + (size_t)(key0 + 64 + (t >> 1)) * 64, 0, 1);
    __syncthreads();

    // scores: 16 q x 64 key. Batch B-fragment loads, then 8 WMMAs.
    v8f sc[4] = {};
    {
      FragB bk[8];
#pragma unroll
      for (int nb = 0; nb < 4; ++nb)
#pragma unroll
        for (int h2 = 0; h2 < 2; ++h2) {
          const __bf16* p = &Kt[(nb * 16 + lane15) * 64 + h2 * 32 + hi * 8];
          bk[nb * 2 + h2].u[0] = *(const uint4*)p;
          bk[nb * 2 + h2].u[1] = *(const uint4*)(p + 16);
        }
#pragma unroll
      for (int nb = 0; nb < 4; ++nb)
#pragma unroll
        for (int h2 = 0; h2 < 2; ++h2)
          sc[nb] = __builtin_amdgcn_wmma_f32_16x16x32_bf16(
              false, aq[h2].v, false, bk[nb * 2 + h2].v, (short)0, sc[nb],
              false, false);
    }

    // online softmax over the 64 new scores per row
    float f[8];
#pragma unroll
    for (int i = 0; i < 8; ++i) {
      float a0 = sc[0][i] * scale, a1 = sc[1][i] * scale;
      float a2 = sc[2][i] * scale, a3 = sc[3][i] * scale;
      float m  = red_max16(fmaxf(fmaxf(a0, a1), fmaxf(a2, a3)));
      float Mn = fmaxf(M[i], m);
      float p0 = __expf(a0 - Mn), p1 = __expf(a1 - Mn);
      float p2 = __expf(a2 - Mn), p3 = __expf(a3 - Mn);
      float ps = red_sum16(p0 + p1 + p2 + p3);
      f[i] = __expf(M[i] - Mn);
      l[i] = l[i] * f[i] + ps;
      M[i] = Mn;
      int row = i + hi * 8;
      Pw[row * 64 + lane15]      = (__bf16)p0;
      Pw[row * 64 + 16 + lane15] = (__bf16)p1;
      Pw[row * 64 + 32 + lane15] = (__bf16)p2;
      Pw[row * 64 + 48 + lane15] = (__bf16)p3;
    }
#pragma unroll
    for (int nb = 0; nb < 4; ++nb)
#pragma unroll
      for (int i = 0; i < 8; ++i) o[nb][i] = o[nb][i] * f[i];

    __syncthreads();  // P staged (C-layout) -> re-read as A-layout

    // P A-fragments (keys [0,32) and [32,64)), V B-fragments, 8 WMMAs.
    FragB ap[2];
#pragma unroll
    for (int kc = 0; kc < 2; ++kc) {
      const __bf16* p = &Pw[lane15 * 64 + kc * 32];
      ap[kc].u[0] = *(const uint4*)(p + hi * 8);
      ap[kc].u[1] = *(const uint4*)(p + 16 + hi * 8);
    }
    FragB bv_[8];
#pragma unroll
    for (int nb = 0; nb < 4; ++nb)
#pragma unroll
      for (int kc = 0; kc < 2; ++kc) {
        const __bf16* p = &Vt[(nb * 16 + lane15) * 64 + kc * 32 + hi * 8];
        bv_[nb * 2 + kc].u[0] = *(const uint4*)p;
        bv_[nb * 2 + kc].u[1] = *(const uint4*)(p + 16);
      }
#pragma unroll
    for (int nb = 0; nb < 4; ++nb)
#pragma unroll
      for (int kc = 0; kc < 2; ++kc)
        o[nb] = __builtin_amdgcn_wmma_f32_16x16x32_bf16(
            false, ap[kc].v, false, bv_[nb * 2 + kc].v, (short)0, o[nb],
            false, false);
  }

  // normalize + write concat [B,S,1024] bf16
  const int bsel = bh >> 4, h = bh & 15;
#pragma unroll
  for (int i = 0; i < 8; ++i) {
    float inv = 1.0f / l[i];
    int srow = q0 + i + hi * 8;
#pragma unroll
    for (int nb = 0; nb < 4; ++nb) {
      int d = nb * 16 + lane15;
      concat[((size_t)bsel * 2048 + srow) * 1024 + h * 64 + d] =
          (__bf16)(o[nb][i] * inv);
    }
  }
}

// ---------------- output projection: concat @ Wo + bo -> fp32 --------------
// grid=(32,16): 256x64 C-tile per block, 2 strips per wave (B-frag reuse).
__global__ __launch_bounds__(256)
void out_gemm(const __bf16* __restrict__ A, const __bf16* __restrict__ Wt,
              const float* __restrict__ bias, float* __restrict__ out) {
  __shared__ __bf16 At[256 * 64];
  __shared__ __bf16 Bt[64 * 64];
  const int t = threadIdx.x;
  const int wv = t >> 5, lane = t & 31, lane15 = lane & 15, hi = lane >> 4;
  const int row0 = blockIdx.x * 256;
  const int n0 = blockIdx.y * 64;

  v8f acc[2][4] = {};
  for (int k0 = 0; k0 < 1024; k0 += 64) {
    __syncthreads();
    {
      const __bf16* g = A + (size_t)(row0 + t) * 1024 + k0;
      __bf16* d = &At[t * 64];
#pragma unroll
      for (int c = 0; c < 64; c += 8) *(uint4*)(d + c) = *(const uint4*)(g + c);
    }
    {
      int r = t >> 2, c = (t & 3) * 16;
      const __bf16* g = Wt + (size_t)(n0 + r) * 1024 + k0 + c;
      __bf16* d = &Bt[r * 64 + c];
      *(uint4*)(d)     = *(const uint4*)(g);
      *(uint4*)(d + 8) = *(const uint4*)(g + 8);
    }
    __syncthreads();

    FragB a[2][2];
#pragma unroll
    for (int s = 0; s < 2; ++s) {
      const int ar = wv * 32 + s * 16 + lane15;
#pragma unroll
      for (int h2 = 0; h2 < 2; ++h2) {
        const __bf16* p = &At[ar * 64 + h2 * 32];
        a[s][h2].u[0] = *(const uint4*)(p + hi * 8);
        a[s][h2].u[1] = *(const uint4*)(p + 16 + hi * 8);
      }
    }
#pragma unroll
    for (int nb = 0; nb < 4; ++nb) {
      FragB b[2];
#pragma unroll
      for (int h2 = 0; h2 < 2; ++h2) {
        const __bf16* p = &Bt[(nb * 16 + lane15) * 64 + h2 * 32 + hi * 8];
        b[h2].u[0] = *(const uint4*)p;
        b[h2].u[1] = *(const uint4*)(p + 16);
      }
#pragma unroll
      for (int s = 0; s < 2; ++s)
#pragma unroll
        for (int h2 = 0; h2 < 2; ++h2)
          acc[s][nb] = __builtin_amdgcn_wmma_f32_16x16x32_bf16(
              false, a[s][h2].v, false, b[h2].v, (short)0, acc[s][nb],
              false, false);
    }
  }
#pragma unroll
  for (int s = 0; s < 2; ++s)
#pragma unroll
    for (int i = 0; i < 8; ++i) {
      int r = row0 + wv * 32 + s * 16 + i + hi * 8;
#pragma unroll
      for (int nb = 0; nb < 4; ++nb) {
        int col = n0 + nb * 16 + lane15;
        out[(size_t)r * 1024 + col] = acc[s][nb][i] + bias[col];
      }
    }
}

// ---------------------------------------------------------------------------

extern "C" void kernel_launch(void* const* d_in, const int* in_sizes, int n_in,
                              void* d_out, int out_size, void* d_ws,
                              size_t ws_size, hipStream_t stream) {
  const float* x  = (const float*)d_in[0];
  const float* Wq = (const float*)d_in[1];
  const float* bq = (const float*)d_in[2];
  const float* Wk = (const float*)d_in[3];
  const float* bk = (const float*)d_in[4];
  const float* Wv = (const float*)d_in[5];
  const float* bv = (const float*)d_in[6];
  const float* Wo = (const float*)d_in[7];
  const float* bo = (const float*)d_in[8];

  char* ws = (char*)d_ws;
  size_t off = 0;
  auto alloc = [&](size_t bytes) {
    char* p = ws + off;
    off += (bytes + 255) & ~(size_t)255;
    return p;
  };
  const size_t ROWS = 8192;  // B*S
  __bf16* xb  = (__bf16*)alloc(ROWS * 1024 * 2);
  __bf16* wqt = (__bf16*)alloc(1024 * 1024 * 2);
  __bf16* wkt = (__bf16*)alloc(1024 * 1024 * 2);
  __bf16* wvt = (__bf16*)alloc(1024 * 1024 * 2);
  __bf16* wot = (__bf16*)alloc(1024 * 1024 * 2);
  __bf16* qb  = (__bf16*)alloc(ROWS * 1024 * 2);   // [B,H,S,D]
  __bf16* kb  = (__bf16*)alloc(ROWS * 1024 * 2);   // [B,H,S,D]
  __bf16* vtb = (__bf16*)alloc(ROWS * 1024 * 2);   // [B,H,D,S]
  __bf16* cc  = (__bf16*)alloc(ROWS * 1024 * 2);   // [B,S,H*D]

  {
    int n = (int)(ROWS * 1024);
    cvt_f32_to_bf16<<<(n / 4 + 255) / 256, 256, 0, stream>>>(x, xb, n);
  }
  dim3 tg(64, 64), tb(16, 16);
  transpose_cvt_bf16<<<tg, tb, 0, stream>>>(Wq, wqt);
  transpose_cvt_bf16<<<tg, tb, 0, stream>>>(Wk, wkt);
  transpose_cvt_bf16<<<tg, tb, 0, stream>>>(Wv, wvt);
  transpose_cvt_bf16<<<tg, tb, 0, stream>>>(Wo, wot);

  dim3 pg(32, 16);
  proj_gemm_ln<<<pg, 256, 0, stream>>>(xb, wqt, bq, qb, 0);
  proj_gemm_ln<<<pg, 256, 0, stream>>>(xb, wkt, bk, kb, 0);
  proj_gemm_ln<<<pg, 256, 0, stream>>>(xb, wvt, bv, vtb, 1);

  flash_attn<<<dim3(16, 64), 256, 0, stream>>>(qb, kb, vtb, cc);

  out_gemm<<<pg, 256, 0, stream>>>(cc, wot, bo, (float*)d_out);
}
